// PointnetSAModuleMSG_att_69561290326465
// MI455X (gfx1250) — compile-verified, matrix operand order checked by reading, not run on hardware
//
#include <hip/hip_runtime.h>
#include <cmath>

// ---------------------------------------------------------------------------
// PointNet++ SA-MSG + attention for MI455X (gfx1250, wave32, WMMA)
// ---------------------------------------------------------------------------

typedef __attribute__((ext_vector_type(16))) _Float16 v16h;
typedef __attribute__((ext_vector_type(8)))  float    v8f;

#define B_   8
#define N_   4096
#define S_   1024
#define NG_  (B_ * S_)      // 8192 groups
#define KATT 640

__device__ __forceinline__ float preluf(float x, float a) { return x >= 0.f ? x : a * x; }
__device__ __forceinline__ float bnf(float x, float g, float b) {
    return x * (g * rsqrtf(1.f + 1e-5f)) + b;
}
// order-preserving float <-> uint encoding (for atomicMax over signed floats)
__device__ __forceinline__ unsigned encF(float f) {
    unsigned u = __float_as_uint(f);
    return (u & 0x80000000u) ? ~u : (u | 0x80000000u);
}
__device__ __forceinline__ float decF(unsigned u) {
    return (u & 0x80000000u) ? __uint_as_float(u & 0x7FFFFFFFu) : __uint_as_float(~u);
}

// ---------------------------------------------------------------------------
// 1) Furthest point sampling: one block per batch. Emits fps_idx and new_xyz.
// ---------------------------------------------------------------------------
__global__ __launch_bounds__(256) void fps_kernel(const float* __restrict__ xyz,
                                                  int* __restrict__ fpsIdx,
                                                  float* __restrict__ newXyz) {
    const int b = blockIdx.x, tid = threadIdx.x;
    const float* X = xyz + (size_t)b * N_ * 3;
    float px[16], py[16], pz[16], md[16];
#pragma unroll
    for (int i = 0; i < 16; ++i) {
        int j = i * 256 + tid;
        px[i] = X[j * 3 + 0]; py[i] = X[j * 3 + 1]; pz[i] = X[j * 3 + 2];
        md[i] = 1e10f;
    }
    __shared__ float sv[256];
    __shared__ int   si[256];
    __shared__ float cx, cy, cz;
    int sel = 0;
    for (int it = 0; it < S_; ++it) {
        if (tid == 0) {
            fpsIdx[b * S_ + it] = sel;
            cx = X[sel * 3 + 0]; cy = X[sel * 3 + 1]; cz = X[sel * 3 + 2];
            float* o = newXyz + ((size_t)b * S_ + it) * 3;
            o[0] = cx; o[1] = cy; o[2] = cz;
        }
        __syncthreads();
        float bx = cx, by = cy, bz = cz;
        float best = -1.f; int bi = 0;
#pragma unroll
        for (int i = 0; i < 16; ++i) {
            int j = i * 256 + tid;
            float dx = px[i] - bx, dy = py[i] - by, dz = pz[i] - bz;
            float d = dx * dx + dy * dy + dz * dz;
            md[i] = fminf(md[i], d);
            if (md[i] > best || (md[i] == best && j < bi)) { best = md[i]; bi = j; }
        }
        sv[tid] = best; si[tid] = bi;
        __syncthreads();
        for (int s = 128; s > 0; s >>= 1) {
            if (tid < s) {
                if (sv[tid + s] > sv[tid] ||
                    (sv[tid + s] == sv[tid] && si[tid + s] < si[tid])) {
                    sv[tid] = sv[tid + s]; si[tid] = si[tid + s];
                }
            }
            __syncthreads();
        }
        sel = si[0];
    }
}

// ---------------------------------------------------------------------------
// 2) Attention per-point MLP 3 -> 64 -> 128 -> 1024, global max via atomicMax
// ---------------------------------------------------------------------------
__global__ __launch_bounds__(256) void init_gmax(unsigned* __restrict__ g) {
    g[blockIdx.x * 256 + threadIdx.x] = 0x00800000u; // encF(-FLT_MAX)
}

__global__ __launch_bounds__(256) void att_point_kernel(
    const float* __restrict__ newXyz,
    const float* __restrict__ c1w, const float* __restrict__ c1b,
    const float* __restrict__ g1,  const float* __restrict__ b1,
    const float* __restrict__ c2w, const float* __restrict__ c2b,
    const float* __restrict__ g2,  const float* __restrict__ b2,
    const float* __restrict__ c3w, const float* __restrict__ c3b,
    const float* __restrict__ g3,  const float* __restrict__ b3,
    const float* __restrict__ preluP, unsigned* __restrict__ gmaxEnc) {
    const int p = blockIdx.x, tid = threadIdx.x, b = p >> 10;
    const float a = preluP[0];
    __shared__ float x[3], h1[64], h2[128];
    if (tid < 3) x[tid] = newXyz[p * 3 + tid];
    __syncthreads();
    if (tid < 64) {
        float s = c1w[tid * 3] * x[0] + c1w[tid * 3 + 1] * x[1] + c1w[tid * 3 + 2] * x[2] + c1b[tid];
        h1[tid] = preluf(bnf(s, g1[tid], b1[tid]), a);
    }
    __syncthreads();
    if (tid < 128) {
        float s = c2b[tid];
        const float* w = c2w + tid * 64;
        for (int k = 0; k < 64; ++k) s += w[k] * h1[k];
        h2[tid] = preluf(bnf(s, g2[tid], b2[tid]), a);
    }
    __syncthreads();
    for (int c = tid; c < 1024; c += 256) {
        float s = c3b[c];
        const float* w = c3w + c * 128;
        for (int k = 0; k < 128; ++k) s += w[k] * h2[k];
        float v = preluf(bnf(s, g3[c], b3[c]), a);
        atomicMax(&gmaxEnc[b * 1024 + c], encF(v));
    }
}

// ---------------------------------------------------------------------------
// 3) Attention FC head: 1024 -> 512 -> 640, sigmoid. One block per batch.
// ---------------------------------------------------------------------------
__global__ __launch_bounds__(256) void att_fc_kernel(
    const unsigned* __restrict__ gmaxEnc,
    const float* __restrict__ fc1w, const float* __restrict__ fc1b,
    const float* __restrict__ g4,   const float* __restrict__ b4,
    const float* __restrict__ fc2w, const float* __restrict__ fc2b,
    const float* __restrict__ preluP, float* __restrict__ att) {
    const int b = blockIdx.x, tid = threadIdx.x;
    const float a = preluP[0];
    __shared__ float g0[1024], h[512];
    for (int c = tid; c < 1024; c += 256) g0[c] = decF(gmaxEnc[b * 1024 + c]);
    __syncthreads();
    for (int o = tid; o < 512; o += 256) {
        float s = fc1b[o];
        const float* w = fc1w + o * 1024;
        for (int k = 0; k < 1024; ++k) s += w[k] * g0[k];
        h[o] = preluf(bnf(s, g4[o], b4[o]), a);
    }
    __syncthreads();
    for (int o = tid; o < KATT; o += 256) {
        float s = fc2b[o];
        const float* w = fc2w + o * 512;
        for (int k = 0; k < 512; ++k) s += w[k] * h[k];
        att[b * KATT + o] = 1.f / (1.f + expf(-s));
    }
}

// ---------------------------------------------------------------------------
// 4) Ball grouping: one wave32 per group; ballot stream-compaction preserves
//    ascending index order; pad with first in-ball index.
// ---------------------------------------------------------------------------
__global__ __launch_bounds__(256) void group_kernel(
    const float* __restrict__ xyz, const float* __restrict__ newXyz,
    int* __restrict__ gi0, int* __restrict__ gi1, int* __restrict__ gi2) {
    const int wave = threadIdx.x >> 5, lane = threadIdx.x & 31;
    const int g = blockIdx.x * 8 + wave;
    const int b = g >> 10;
    const float* X = xyz + (size_t)b * N_ * 3;
    const float cx = newXyz[g * 3], cy = newXyz[g * 3 + 1], cz = newXyz[g * 3 + 2];
    const float r2s[3] = {0.01f, 0.04f, 0.16f};
    const int   nss[3] = {16, 32, 64};
    int* outs[3] = {gi0 + g * 16, gi1 + g * 32, gi2 + g * 64};
    for (int br = 0; br < 3; ++br) {
        const int ns = nss[br];
        const float r2 = r2s[br];
        int* og = outs[br];
        int cnt = 0, first = -1;
        for (int base = 0; base < N_ && cnt < ns; base += 32) {
            int j = base + lane;
            float dx = X[j * 3] - cx, dy = X[j * 3 + 1] - cy, dz = X[j * 3 + 2] - cz;
            bool pred = (dx * dx + dy * dy + dz * dz) < r2;
            unsigned m = (unsigned)__ballot(pred);
            int pos = cnt + __popc(m & ((1u << lane) - 1u));
            if (pred && pos < ns) { og[pos] = j; if (pos == 0) first = j; }
            cnt += __popc(m);
        }
#pragma unroll
        for (int off = 16; off > 0; off >>= 1) first = max(first, __shfl_xor(first, off));
        for (int p = cnt + lane; p < ns; p += 32) og[p] = first;
    }
}

// ---------------------------------------------------------------------------
// 5) Fused shared-MLP branch: gather -> 3x (WMMA GEMM + BN + ReLU) ->
//    neighbor max-pool -> attention scale -> (B,640,S) output.
//    All weights + activations resident in the 320KB CDNA5 LDS.
// ---------------------------------------------------------------------------
template <int MT, int Cout, int KT>
__device__ __forceinline__ void wmma_layer(const _Float16* __restrict__ Xl, int xs,
                                           const _Float16* __restrict__ Wl, int ws_,
                                           const float* __restrict__ scale,
                                           const float* __restrict__ bias,
                                           _Float16* __restrict__ Ol, int os, int tid) {
    constexpr int NT = Cout / 16;
    const int wave = tid >> 5, lane = tid & 31;
    const int r = lane & 15, hi = lane >> 4;
    for (int t = wave; t < MT * NT; t += 8) {
        const int mt = t / NT, nt = t % NT;
        v8f acc = {};
#pragma unroll
        for (int kk = 0; kk < KT; kk += 32) {
            v16h av, bv;
            const _Float16* ap = Xl + (mt * 16 + r) * xs + kk + hi * 8;
            const _Float16* bp = Wl + (nt * 16 + r) * ws_ + kk + hi * 8;
#pragma unroll
            for (int i = 0; i < 8; ++i) {
                av[i] = ap[i]; av[8 + i] = ap[16 + i];
                bv[i] = bp[i]; bv[8 + i] = bp[16 + i];
            }
            acc = __builtin_amdgcn_wmma_f32_16x16x32_f16(false, av, false, bv,
                                                         (short)0, acc, false, false);
        }
        const int col = nt * 16 + r;
        const float sc = scale[col], bi = bias[col];
#pragma unroll
        for (int q = 0; q < 8; ++q) {
            float v = fmaxf(acc[q] * sc + bi, 0.f);
            Ol[(mt * 16 + q + 8 * hi) * os + col] = (_Float16)v;
        }
    }
}

template <int MT, int Cout, int KT>
__device__ __forceinline__ void wmma_layer_max(const _Float16* __restrict__ Xl, int xs,
                                               const _Float16* __restrict__ Wl, int ws_,
                                               const float* __restrict__ scale,
                                               const float* __restrict__ bias,
                                               unsigned* __restrict__ sMax, int tid) {
    constexpr int NT = Cout / 16;
    const int wave = tid >> 5, lane = tid & 31;
    const int r = lane & 15, hi = lane >> 4;
    for (int t = wave; t < MT * NT; t += 8) {
        const int mt = t / NT, nt = t % NT;
        v8f acc = {};
#pragma unroll
        for (int kk = 0; kk < KT; kk += 32) {
            v16h av, bv;
            const _Float16* ap = Xl + (mt * 16 + r) * xs + kk + hi * 8;
            const _Float16* bp = Wl + (nt * 16 + r) * ws_ + kk + hi * 8;
#pragma unroll
            for (int i = 0; i < 8; ++i) {
                av[i] = ap[i]; av[8 + i] = ap[16 + i];
                bv[i] = bp[i]; bv[8 + i] = bp[16 + i];
            }
            acc = __builtin_amdgcn_wmma_f32_16x16x32_f16(false, av, false, bv,
                                                         (short)0, acc, false, false);
        }
        const int col = nt * 16 + r;
        const float sc = scale[col], bi = bias[col];
        float m = 0.f; // post-ReLU values are >= 0
#pragma unroll
        for (int q = 0; q < 8; ++q) m = fmaxf(m, fmaxf(acc[q] * sc + bi, 0.f));
        atomicMax(&sMax[col], __float_as_uint(m)); // bits order == value order for >=0
    }
}

template <int NS, int C1, int C2, int C3, int COFF>
__global__ __launch_bounds__(256) void mlp_kernel(
    const float* __restrict__ xyz, const float* __restrict__ features,
    const float* __restrict__ newXyz, const int* __restrict__ gidx,
    const float* __restrict__ w1, const float* __restrict__ g1, const float* __restrict__ b1,
    const float* __restrict__ w2, const float* __restrict__ g2, const float* __restrict__ b2,
    const float* __restrict__ w3, const float* __restrict__ g3, const float* __restrict__ b3,
    const float* __restrict__ att, float* __restrict__ outFeats) {
    constexpr int MT = NS / 16;
    constexpr int XS = 96 + 8, W1S = 96 + 8, W2S = C1 + 8, W3S = C2 + 8;
    constexpr int A1S = C1 + 8, A2S = C2 + 8;
    __shared__ _Float16 sW1[C1 * W1S];
    __shared__ _Float16 sW2[C2 * W2S];
    __shared__ _Float16 sW3[C3 * W3S];
    __shared__ _Float16 sX[NS * XS];
    __shared__ _Float16 sA1[NS * A1S];
    __shared__ _Float16 sA2[NS * A2S];
    __shared__ float sS1[C1], sB1[C1], sS2[C2], sB2[C2], sS3[C3], sB3[C3];
    __shared__ unsigned sMax[C3];
    __shared__ int sIdx[NS];
    const int tid = threadIdx.x;

    // ---- load weights (f32 -> f16, K padded with zeros for layer 1) ----
    for (int i = tid; i < C1 * 96; i += 256) {
        int r = i / 96, c = i % 96;
        sW1[r * W1S + c] = (c < 67) ? (_Float16)w1[r * 67 + c] : (_Float16)0.f;
    }
    for (int i = tid; i < C2 * C1; i += 256)
        sW2[(i / C1) * W2S + (i % C1)] = (_Float16)w2[i];
    for (int i = tid; i < C3 * C2; i += 256)
        sW3[(i / C2) * W3S + (i % C2)] = (_Float16)w3[i];
    const float inv = rsqrtf(1.f + 1e-5f);
    for (int i = tid; i < C1; i += 256) { sS1[i] = g1[i] * inv; sB1[i] = b1[i]; }
    for (int i = tid; i < C2; i += 256) { sS2[i] = g2[i] * inv; sB2[i] = b2[i]; }
    for (int i = tid; i < C3; i += 256) { sS3[i] = g3[i] * inv; sB3[i] = b3[i]; }

    for (int g = blockIdx.x; g < NG_; g += gridDim.x) {
        __syncthreads();
        const int b = g >> 10, s = g & 1023;
        if (tid < NS) sIdx[tid] = gidx[g * NS + tid];
        for (int c = tid; c < C3; c += 256) sMax[c] = 0u;
        __syncthreads();
        // ---- gather neighbors: [rel-xyz | features | zero-pad] -> (NS x 96) f16
        for (int i = tid; i < NS * 96; i += 256) {
            int row = i / 96, c = i % 96;
            int j = sIdx[row];
            float v;
            if (c < 3)       v = xyz[((size_t)b * N_ + j) * 3 + c] - newXyz[g * 3 + c];
            else if (c < 67) v = features[((size_t)b * N_ + j) * 64 + (c - 3)];
            else             v = 0.f;
            sX[row * XS + c] = (_Float16)v;
        }
        __syncthreads();
        wmma_layer<MT, C1, 96>(sX, XS, sW1, W1S, sS1, sB1, sA1, A1S, tid);
        __syncthreads();
        wmma_layer<MT, C2, C1>(sA1, A1S, sW2, W2S, sS2, sB2, sA2, A2S, tid);
        __syncthreads();
        wmma_layer_max<MT, C3, C2>(sA2, A2S, sW3, W3S, sS3, sB3, sMax, tid);
        __syncthreads();
        for (int c = tid; c < C3; c += 256) {
            float v = __uint_as_float(sMax[c]);
            outFeats[((size_t)b * KATT + COFF + c) * S_ + s] = v * att[b * KATT + COFF + c];
        }
    }
}

// ---------------------------------------------------------------------------
// Host launcher
// ---------------------------------------------------------------------------
extern "C" void kernel_launch(void* const* d_in, const int* in_sizes, int n_in,
                              void* d_out, int out_size, void* d_ws, size_t ws_size,
                              hipStream_t stream) {
    (void)in_sizes; (void)n_in; (void)out_size; (void)ws_size;
    const float* xyz      = (const float*)d_in[0];
    const float* features = (const float*)d_in[1];
    // attention params (dict order)
    const float* c1w = (const float*)d_in[2];  const float* c1b = (const float*)d_in[3];
    const float* bg1 = (const float*)d_in[4];  const float* bb1 = (const float*)d_in[5];
    const float* c2w = (const float*)d_in[6];  const float* c2b = (const float*)d_in[7];
    const float* bg2 = (const float*)d_in[8];  const float* bb2 = (const float*)d_in[9];
    const float* c3w = (const float*)d_in[10]; const float* c3b = (const float*)d_in[11];
    const float* bg3 = (const float*)d_in[12]; const float* bb3 = (const float*)d_in[13];
    const float* f1w = (const float*)d_in[14]; const float* f1b = (const float*)d_in[15];
    const float* bg4 = (const float*)d_in[16]; const float* bb4 = (const float*)d_in[17];
    const float* f2w = (const float*)d_in[18]; const float* f2b = (const float*)d_in[19];
    const float* prl = (const float*)d_in[20];
    // mlps: 3 branches x 3 layers x (w,g,b) starting at index 21
    const float* mp[27];
    for (int i = 0; i < 27; ++i) mp[i] = (const float*)d_in[21 + i];

    float* newXyz = (float*)d_out;                    // (B,1024,3)
    float* feats  = (float*)d_out + B_ * S_ * 3;      // (B,640,1024)

    char* ws = (char*)d_ws;
    int*      fpsIdx  = (int*)ws;                       // 8192 ints
    unsigned* gmaxEnc = (unsigned*)(ws + 32768);        // 8192 uints
    float*    attv    = (float*)(ws + 65536);           // B*640 floats
    int*      gi0     = (int*)(ws + 65536 + 20480);     // 8192*16
    int*      gi1     = gi0 + NG_ * 16;                 // 8192*32
    int*      gi2     = gi1 + NG_ * 32;                 // 8192*64

    fps_kernel<<<B_, 256, 0, stream>>>(xyz, fpsIdx, newXyz);
    init_gmax<<<(B_ * 1024) / 256, 256, 0, stream>>>(gmaxEnc);
    att_point_kernel<<<NG_, 256, 0, stream>>>(newXyz, c1w, c1b, bg1, bb1,
                                              c2w, c2b, bg2, bb2,
                                              c3w, c3b, bg3, bb3, prl, gmaxEnc);
    att_fc_kernel<<<B_, 256, 0, stream>>>(gmaxEnc, f1w, f1b, bg4, bb4, f2w, f2b, prl, attv);
    group_kernel<<<NG_ / 8, 256, 0, stream>>>(xyz, newXyz, gi0, gi1, gi2);

    mlp_kernel<16, 64, 64, 128, 0><<<1024, 256, 0, stream>>>(
        xyz, features, newXyz, gi0,
        mp[0], mp[1], mp[2], mp[3], mp[4], mp[5], mp[6], mp[7], mp[8], attv, feats);
    mlp_kernel<32, 128, 128, 256, 128><<<1024, 256, 0, stream>>>(
        xyz, features, newXyz, gi1,
        mp[9], mp[10], mp[11], mp[12], mp[13], mp[14], mp[15], mp[16], mp[17], attv, feats);
    mlp_kernel<64, 128, 128, 256, 384><<<1024, 256, 0, stream>>>(
        xyz, features, newXyz, gi2,
        mp[18], mp[19], mp[20], mp[21], mp[22], mp[23], mp[24], mp[25], mp[26], attv, feats);
}
